// LightweightReservoir_10582799417574
// MI455X (gfx1250) — compile-verified, matrix operand order checked by reading, not run on hardware
//
#include <hip/hip_runtime.h>
#include <hip/hip_bf16.h>

#define BATCH 64
#define SEQ   4096
#define ISZ   64
#define RSZ   128
#define LEAK  0.5f

typedef __attribute__((ext_vector_type(16))) _Float16 v16h;
typedef __attribute__((ext_vector_type(8)))  _Float16 v8h;
typedef __attribute__((ext_vector_type(4)))  _Float16 v4h;
typedef __attribute__((ext_vector_type(8)))  float    v8f;
typedef __attribute__((ext_vector_type(4)))  float    v4f;

// CDNA5 transcendental tanh (VOP1 v_tanh_f32). TRANS ops need one independent
// instruction before the result is consumed -> trailing v_nop for safety.
__device__ __forceinline__ float tanh_hw(float x) {
  float y;
  asm("v_tanh_f32 %0, %1\n\tv_nop" : "=v"(y) : "v"(x));
  return y;
}

// Gather one 16x32 f16 WMMA fragment (A or B) from an LDS image stored as
// [row][k] row-major halves. Caller passes the row pointer (row already
// includes lane&15) and colbase = kc*32 + 8*(lane>>4).
// Lane/half -> K mapping: halves 0..7 -> K = colbase+0..7,
// halves 8..15 -> K = colbase+16..23  (two ds_load_b128 per fragment).
__device__ __forceinline__ v16h frag_ld(const _Float16* rowp, int col) {
  v8h lo = *(const v8h*)(rowp + col);
  v8h hi = *(const v8h*)(rowp + col + 16);
  v16h r;
#pragma unroll
  for (int i = 0; i < 8; ++i) { r[i] = lo[i]; r[i + 8] = hi[i]; }
  return r;
}

// ---------------------------------------------------------------------------
// Kernel 1: U[b*S+t, r] = tanh( X[b*S+t, :] . W_in[r, :] )   (M=262144,N=128,K=64)
// One workgroup = 8 waves = 8 M-tiles of 16 rows. f16 WMMA, f32 accumulate.
// ---------------------------------------------------------------------------
__global__ __launch_bounds__(256) void esn_input_proj(const float* __restrict__ X,
                                                      const float* __restrict__ Win,
                                                      float* __restrict__ U) {
  __shared__ _Float16 sW[RSZ * ISZ];   // W_in as f16, [r][i]  (16 KB)
  __shared__ _Float16 sX[128 * ISZ];   // X block as f16, [m][i] (16 KB)

  const int tid  = threadIdx.x;
  const int lane = tid & 31;
  const int wv   = tid >> 5;
  const long m0  = (long)blockIdx.x * 128;

  // Stage W_in (f32 -> f16)
  for (int idx = tid; idx < (RSZ * ISZ) / 4; idx += 256) {
    float4 f = ((const float4*)Win)[idx];
    v4h h = { (_Float16)f.x, (_Float16)f.y, (_Float16)f.z, (_Float16)f.w };
    *(v4h*)&sW[idx * 4] = h;
  }
  // Stage X block (f32 -> f16)
  const float* Xb = X + m0 * ISZ;
  for (int idx = tid; idx < (128 * ISZ) / 4; idx += 256) {
    float4 f = ((const float4*)Xb)[idx];
    v4h h = { (_Float16)f.x, (_Float16)f.y, (_Float16)f.z, (_Float16)f.w };
    *(v4h*)&sX[idx * 4] = h;
  }
  __syncthreads();

  const int l15 = lane & 15;
  const int cb  = 8 * (lane >> 4);

  // A fragments: rows of X tile (K = 64 -> 2 chunks), reused across all N tiles
  const _Float16* arow = &sX[(16 * wv + l15) * ISZ];
  v16h a0 = frag_ld(arow, 0 + cb);
  v16h a1 = frag_ld(arow, 32 + cb);

  const int mloc = 8 * (lane >> 4);
#pragma unroll
  for (int nt = 0; nt < 8; ++nt) {
    // B[k][n] = W_in[16*nt + n][k] -> same gather pattern on sW rows
    const _Float16* brow = &sW[(16 * nt + l15) * ISZ];
    v16h b0 = frag_ld(brow, 0 + cb);
    v16h b1 = frag_ld(brow, 32 + cb);
    v8f c = {};
    c = __builtin_amdgcn_wmma_f32_16x16x32_f16(false, a0, false, b0, (short)0, c, false, false);
    c = __builtin_amdgcn_wmma_f32_16x16x32_f16(false, a1, false, b1, (short)0, c, false, false);
    // D layout: lane<16 -> N=lane, M=v ; lane>=16 -> N=lane-16, M=v+8
#pragma unroll
    for (int v = 0; v < 8; ++v) {
      long row = m0 + 16 * wv + mloc + v;
      U[row * RSZ + 16 * nt + l15] = tanh_hw(c[v]);
    }
  }
}

// ---------------------------------------------------------------------------
// Kernel 2: sequential reservoir scan, transposed form H_new^T = W_res x H^T.
// grid = 4 blocks (16 batches each), 8 waves; wave mt owns R rows 16mt..16mt+15.
// A = W_res tile lives in VGPRs all 4096 steps; B = H^T re-gathered from a
// double-buffered f16 LDS image each step. u_t read from d_out (written by
// kernel 1) and overwritten in place with states_t.
// ---------------------------------------------------------------------------
__global__ __launch_bounds__(256) void esn_scan(const float* __restrict__ Wres,
                                                float* __restrict__ UY) {
  __shared__ _Float16 sW[RSZ * RSZ];      // W_res f16, [r][k]     (32 KB)
  __shared__ _Float16 sH[2][16 * RSZ];    // H f16, [buf][b][r]    ( 8 KB)

  const int tid  = threadIdx.x;
  const int lane = tid & 31;
  const int mt   = tid >> 5;     // wave id = R row-tile
  const int bg   = blockIdx.x;   // batch group of 16

  // Stage W_res (f32 -> f16)
  for (int idx = tid; idx < (RSZ * RSZ) / 4; idx += 256) {
    float4 f = ((const float4*)Wres)[idx];
    v4h h = { (_Float16)f.x, (_Float16)f.y, (_Float16)f.z, (_Float16)f.w };
    *(v4h*)&sW[idx * 4] = h;
  }
  // h_0 = 0
  {
    v4h z = { (_Float16)0.f, (_Float16)0.f, (_Float16)0.f, (_Float16)0.f };
    for (int idx = tid; idx < (16 * RSZ) / 4; idx += 256)
      *(v4h*)&sH[0][idx * 4] = z;
  }
  __syncthreads();

  const int l15 = lane & 15;
  const int cb  = 8 * (lane >> 4);

  // A = W_res fragments: constant for the whole scan (4 K-chunks)
  v16h aw[4];
#pragma unroll
  for (int kc = 0; kc < 4; ++kc)
    aw[kc] = frag_ld(&sW[(16 * mt + l15) * RSZ], kc * 32 + cb);

  // D layout (transposed product): lane<16 -> batch=lane, r=16mt+v ;
  // lane>=16 -> batch=lane-16, r=16mt+8+v.  Per lane: 8 contiguous r values.
  const int b  = bg * 16 + l15;
  const int r0 = 16 * mt + cb;
  float* base = UY + (size_t)b * SEQ * RSZ + r0;

  v8f h = {};                               // f32 state slice h_t[b][r0..r0+7]
  v4f u_lo = *(const v4f*)(base + 0);       // u_0
  v4f u_hi = *(const v4f*)(base + 4);

  int p = 0;
  for (int t = 0; t < SEQ; ++t) {
    float* rowp = base + (size_t)t * RSZ;

    // Prefetch u_{t+1} early to hide global latency behind the WMMA chain
    v4f un_lo = {}, un_hi = {};
    if (t + 1 < SEQ) {
      un_lo = *(const v4f*)(rowp + RSZ);
      un_hi = *(const v4f*)(rowp + RSZ + 4);
    }

    // (h_t W_res^T)^T tile: 4 chained f16 WMMAs, B = H^T from sH[p]
    v8f c = {};
#pragma unroll
    for (int kc = 0; kc < 4; ++kc) {
      v16h bf = frag_ld(&sH[p][l15 * RSZ], kc * 32 + cb);
      c = __builtin_amdgcn_wmma_f32_16x16x32_f16(false, aw[kc], false, bf, (short)0, c, false, false);
    }

    // Leaky update in f32; emit f16 copy for next step's B fragments
    v8h hh;
    v4f o_lo, o_hi;
#pragma unroll
    for (int v = 0; v < 8; ++v) {
      float uv = (v < 4) ? u_lo[v] : u_hi[v - 4];
      float hn = (1.0f - LEAK) * h[v] + LEAK * tanh_hw(uv + c[v]);
      h[v] = hn;
      hh[v] = (_Float16)hn;
      if (v < 4) o_lo[v] = hn; else o_hi[v - 4] = hn;
    }
    *(v4f*)(rowp + 0) = o_lo;               // states[b][t][r0..r0+3]
    *(v4f*)(rowp + 4) = o_hi;               // states[b][t][r0+4..r0+7]
    *(v8h*)&sH[p ^ 1][l15 * RSZ + r0] = hh; // publish h_{t+1} (f16)

    u_lo = un_lo; u_hi = un_hi;
    __syncthreads();                        // one barrier/step (double buffer)
    p ^= 1;
  }
}

extern "C" void kernel_launch(void* const* d_in, const int* in_sizes, int n_in,
                              void* d_out, int out_size, void* d_ws, size_t ws_size,
                              hipStream_t stream) {
  const float* X    = (const float*)d_in[0];   // [64,4096,64]
  const float* Win  = (const float*)d_in[1];   // [128,64]
  const float* Wres = (const float*)d_in[2];   // [128,128]
  float* out = (float*)d_out;                  // [64,4096,128]
  (void)in_sizes; (void)n_in; (void)out_size; (void)d_ws; (void)ws_size;

  esn_input_proj<<<(BATCH * SEQ) / 128, 256, 0, stream>>>(X, Win, out);
  esn_scan<<<BATCH / 16, 256, 0, stream>>>(Wres, out);
}
